// DctCompress_77970836292151
// MI455X (gfx1250) — matrix operand order repaired and checked by reading, not compiled, not run on hardware
//
#include <hip/hip_runtime.h>

// ---------------- types for WMMA ----------------
typedef _Float16 v16h __attribute__((ext_vector_type(16)));
typedef float    v8f  __attribute__((ext_vector_type(8)));

// ---------------- problem constants ----------------
#define HH 768
#define WW 768
#define HB 96
#define WB 96
#define HWB (HB*WB)                 // 9216 blocks per plane
#define NBATCH 8
#define NPLANES (NBATCH*3)          // 24
#define SLEN (192*HWB)              // 1,769,472 elements per sample stream
#define LP1 (SLEN+1)
#define CHUNK 1024
#define NCHUNKS_PER_S (SLEN/CHUNK)  // 1728
#define NCHUNKS (NBATCH*NCHUNKS_PER_S) // 13824
#define VAL_OFF_ 1027
#define IDX_OFF_ 3
#define EOS_ 2

// Q=50 => scale s=100 => clip(floor((t*100+50)/100),1,255) == t for these integer tables.
// Stored as reciprocals so the quantize epilogue is a single v_mul + v_rndne.
__constant__ float RQL[64] = {
  1.f/16,1.f/11,1.f/10,1.f/16,1.f/24,1.f/40,1.f/51,1.f/61,
  1.f/12,1.f/12,1.f/14,1.f/19,1.f/26,1.f/58,1.f/60,1.f/55,
  1.f/14,1.f/13,1.f/16,1.f/24,1.f/40,1.f/57,1.f/69,1.f/56,
  1.f/14,1.f/17,1.f/22,1.f/29,1.f/51,1.f/87,1.f/80,1.f/62,
  1.f/18,1.f/22,1.f/37,1.f/56,1.f/68,1.f/109,1.f/103,1.f/77,
  1.f/24,1.f/35,1.f/55,1.f/64,1.f/81,1.f/104,1.f/113,1.f/92,
  1.f/49,1.f/64,1.f/78,1.f/87,1.f/103,1.f/121,1.f/120,1.f/101,
  1.f/72,1.f/92,1.f/95,1.f/98,1.f/112,1.f/100,1.f/103,1.f/99 };
__constant__ float RQC[64] = {
  1.f/17,1.f/18,1.f/24,1.f/47,1.f/99,1.f/99,1.f/99,1.f/99,
  1.f/18,1.f/21,1.f/26,1.f/66,1.f/99,1.f/99,1.f/99,1.f/99,
  1.f/24,1.f/26,1.f/56,1.f/99,1.f/99,1.f/99,1.f/99,1.f/99,
  1.f/47,1.f/66,1.f/99,1.f/99,1.f/99,1.f/99,1.f/99,1.f/99,
  1.f/99,1.f/99,1.f/99,1.f/99,1.f/99,1.f/99,1.f/99,1.f/99,
  1.f/99,1.f/99,1.f/99,1.f/99,1.f/99,1.f/99,1.f/99,1.f/99,
  1.f/99,1.f/99,1.f/99,1.f/99,1.f/99,1.f/99,1.f/99,1.f/99,
  1.f/99,1.f/99,1.f/99,1.f/99,1.f/99,1.f/99,1.f/99,1.f/99 };
// ZOFF3[m] = zigzag_rank(m) * 3 * HWB  (stream-row offset for coeff raster index m)
#define Z3(r) ((r)*3*HWB)
__constant__ int ZOFF3[64] = {
  Z3(0), Z3(1), Z3(5), Z3(6), Z3(14),Z3(15),Z3(27),Z3(28),
  Z3(2), Z3(4), Z3(7), Z3(13),Z3(16),Z3(26),Z3(29),Z3(42),
  Z3(3), Z3(8), Z3(12),Z3(17),Z3(25),Z3(30),Z3(41),Z3(43),
  Z3(9), Z3(11),Z3(18),Z3(24),Z3(31),Z3(40),Z3(44),Z3(53),
  Z3(10),Z3(19),Z3(23),Z3(32),Z3(39),Z3(45),Z3(52),Z3(54),
  Z3(20),Z3(22),Z3(33),Z3(38),Z3(46),Z3(51),Z3(55),Z3(60),
  Z3(21),Z3(34),Z3(37),Z3(47),Z3(50),Z3(56),Z3(59),Z3(61),
  Z3(35),Z3(36),Z3(48),Z3(49),Z3(57),Z3(58),Z3(62),Z3(63) };

// ---------------- K0: DCT basis in exact f16 A-fragment layout ----------------
// AFRAG[((mtile*2+kstep)*32 + lane)*16 + j]  (contiguous-K A 16x32 layout matching B:
//   K = j + (lane>=16 ? 16 : 0), M = mtile*16 + (lane&15))
__global__ void k_init_afrag(_Float16* __restrict__ afrag) {
  int t = threadIdx.x;                       // 256 slots exactly
  int mtile = t >> 6, kstep = (t >> 5) & 1, lane = t & 31;
  int m = mtile * 16 + (lane & 15);
  int u = m >> 3, v = m & 7;
  float alu = u ? 1.0f : 0.70710678118654752f;
  float alv = v ? 1.0f : 0.70710678118654752f;
  const float PI8 = 0.39269908169872414f;    // pi/8
  for (int j = 0; j < 16; ++j) {
    int jm = j + ((lane & 16) ? 16 : 0);
    int kp = kstep * 32 + jm;
    int p = kp >> 3, q = kp & 7;
    float w = alu * alv * 0.25f * cosf((q + 0.5f) * v * PI8)
                                * cosf((p + 0.5f) * u * PI8);
    afrag[t * 16 + j] = (_Float16)w;
  }
}

// ---------------- K1: color transform + chroma 2x down/up, write f16 blocks ----------------
// blocks layout: [plane=b*3+c][block=hb*96+wb][k=p*8+q]  (k contiguous -> B-fragment loads)
__global__ void k_color(const float* __restrict__ x, _Float16* __restrict__ blocks) {
  int tid = blockIdx.x * 256 + threadIdx.x;  // 8*384*384 = 1,179,648 exact
  int b  = tid / (384 * 384);
  int r  = tid - b * (384 * 384);
  int h2 = r / 384, w2 = r - (r / 384) * 384;
  int h = h2 * 2, w = w2 * 2;
  const float* xb = x + (size_t)b * 3 * HH * WW;
  float Y[4], cb = 0.f, cr = 0.f;
  #pragma unroll
  for (int i = 0; i < 4; ++i) {
    int hh = h + (i >> 1), ww = w + (i & 1);
    size_t p = (size_t)hh * WW + ww;
    float R = xb[p], G = xb[(size_t)HH * WW + p], Bv = xb[2 * (size_t)HH * WW + p];
    Y[i] = 0.299f * R + 0.587f * G + 0.114f * Bv;
    cb += -0.168736f * R - 0.331264f * G + 0.5f      * Bv;
    cr +=  0.5f      * R - 0.418688f * G - 0.081312f * Bv;
  }
  cb = cb * 0.25f + 0.5f - 128.0f;   // avg + COLOR_B bias - 128
  cr = cr * 0.25f + 0.5f - 128.0f;
  size_t pl0 = (size_t)(b * 3) * HWB * 64;
  #pragma unroll
  for (int i = 0; i < 4; ++i) {
    int hh = h + (i >> 1), ww = w + (i & 1);
    size_t blk = (size_t)((hh >> 3) * WB + (ww >> 3)) * 64 + (hh & 7) * 8 + (ww & 7);
    blocks[pl0 + blk]                        = (_Float16)(Y[i] - 128.0f);
    blocks[pl0 + (size_t)HWB * 64 + blk]     = (_Float16)cb;
    blocks[pl0 + 2 * (size_t)HWB * 64 + blk] = (_Float16)cr;
  }
}

// ---------------- K2: blockwise DCT as 64x64x(N) GEMM via v_wmma_f32_16x16x32_f16 ----------------
// Fused epilogue: quantize (round-half-even via reciprocal mul), zigzag, interleave
// into per-sample stream zs[b][(n_zz*3+ch)*9216 + block] using pure 32-bit index math.
__global__ void k_dct_wmma(const _Float16* __restrict__ afrag,
                           const _Float16* __restrict__ blocks,
                           int* __restrict__ zs) {
  int plane = blockIdx.x / 288;              // 24 planes * 288 blocks = 6912
  int tg    = blockIdx.x - plane * 288;
  int wave  = threadIdx.x >> 5;              // 8 waves (wave32)
  int lane  = threadIdx.x & 31;
  int mtile = wave & 3;
  int ntile = tg * 2 + (wave >> 2);
  int b = plane / 3, ch = plane - b * 3;

  // A fragments: one contiguous 32B load each (pre-swizzled into VGPR layout)
  const v16h* A = (const v16h*)afrag;
  v16h a0 = A[(mtile * 2 + 0) * 32 + lane];
  v16h a1 = A[(mtile * 2 + 1) * 32 + lane];

  // B fragments: 32x16 K-major per lane; lanes 0-15 <-> K 0..15, 16-31 <-> K 16..31
  int nblk = ntile * 16 + (lane & 15);
  const _Float16* bp = blocks + ((size_t)plane * HWB + nblk) * 64 + ((lane >> 4) & 1) * 16;
  v16h b0 = *(const v16h*)(bp);        // K  0..31 slice
  v16h b1 = *(const v16h*)(bp + 32);   // K 32..63 slice

  v8f c = {};
  c = __builtin_amdgcn_wmma_f32_16x16x32_f16(false, a0, false, b0, (short)0, c, false, false);
  c = __builtin_amdgcn_wmma_f32_16x16x32_f16(false, a1, false, b1, (short)0, c, false, false);

  // D layout: VGPR r -> M = r (+8 for lanes 16-31), N = lane&15
  int mhi = (lane >> 4) << 3;
  int base = b * (192 * HWB) + ch * HWB + nblk;   // < 2^31, per-thread uniform
  #pragma unroll
  for (int r = 0; r < 8; ++r) {
    int m = mtile * 16 + r + mhi;
    float rq = (ch == 0) ? RQL[m] : RQC[m];
    int val = (int)rintf(c[r] * rq);         // jnp.round == round-half-even
    zs[base + ZOFF3[m]] = val;
  }
}

// ---------------- K3: per-1024-chunk nonzero counts ----------------
__global__ void k_count(const int* __restrict__ zs, int* __restrict__ counts) {
  __shared__ int sd[256];
  size_t base = (size_t)blockIdx.x * CHUNK + threadIdx.x * 4;
  int4 v = *(const int4*)(zs + base);
  int c = (v.x != 0) + (v.y != 0) + (v.z != 0) + (v.w != 0);
  sd[threadIdx.x] = c; __syncthreads();
  for (int s = 128; s > 0; s >>= 1) {
    if (threadIdx.x < s) sd[threadIdx.x] += sd[threadIdx.x + s];
    __syncthreads();
  }
  if (threadIdx.x == 0) counts[blockIdx.x] = sd[0];
}

// ---------------- K4: per-sample exclusive scan over 1728 chunk counts ----------------
__global__ void k_scan(const int* __restrict__ counts, int* __restrict__ offs,
                       int* __restrict__ nnz) {
  __shared__ int sd[256];
  int s = blockIdx.x, t = threadIdx.x;
  int carry = 0;
  for (int base = 0; base < NCHUNKS_PER_S; base += 256) {
    int i = base + t;
    int v = (i < NCHUNKS_PER_S) ? counts[s * NCHUNKS_PER_S + i] : 0;
    sd[t] = v; __syncthreads();
    for (int off = 1; off < 256; off <<= 1) {
      int tmp = (t >= off) ? sd[t - off] : 0;
      __syncthreads();
      sd[t] += tmp;
      __syncthreads();
    }
    if (i < NCHUNKS_PER_S) offs[s * NCHUNKS_PER_S + i] = carry + sd[t] - v;
    __syncthreads();
    carry += sd[255];
    __syncthreads();
  }
  if (t == 0) nnz[s] = carry;
}

// ---------------- K5: zero-fill output with B128 stores ----------------
__global__ void k_fill(int* __restrict__ out, int n4) {
  int i = blockIdx.x * 256 + threadIdx.x;
  if (i < n4) ((int4*)out)[i] = make_int4(0, 0, 0, 0);
}

// ---------------- K6: stable scatter of nonzeros ----------------
__global__ void k_scatter(const int* __restrict__ zs, const int* __restrict__ offs,
                          int* __restrict__ out) {
  __shared__ int sd[256];
  int cid = blockIdx.x;
  int s = cid / NCHUNKS_PER_S;
  int cis = cid - s * NCHUNKS_PER_S;
  int t = threadIdx.x;
  size_t gbase = (size_t)cid * CHUNK + t * 4;
  int4 v = *(const int4*)(zs + gbase);
  int vv[4] = { v.x, v.y, v.z, v.w };
  int cnt = (v.x != 0) + (v.y != 0) + (v.z != 0) + (v.w != 0);
  sd[t] = cnt; __syncthreads();
  for (int off = 1; off < 256; off <<= 1) {
    int tmp = (t >= off) ? sd[t - off] : 0;
    __syncthreads();
    sd[t] += tmp;
    __syncthreads();
  }
  int pos = offs[cid] + sd[t] - cnt;         // chunk offset + intra-chunk stable prefix
  int flbase = cis * CHUNK + t * 4;
  int* ob = out + (size_t)s * 3 * LP1;
  #pragma unroll
  for (int i = 0; i < 4; ++i) {
    int z = vv[i];
    if (z != 0) {
      int fl = flbase + i;
      int row = fl / HWB;
      int col = fl - row * HWB;
      ob[pos]                     = row + IDX_OFF_;
      ob[(size_t)LP1 + pos]       = col + IDX_OFF_;
      ob[2 * (size_t)LP1 + pos]   = z + VAL_OFF_;
      ++pos;
    }
  }
}

// ---------------- K7: EOS markers ----------------
__global__ void k_eos(const int* __restrict__ nnz, int* __restrict__ out) {
  int t = threadIdx.x;
  if (t >= 24) return;
  int s = t / 3, r = t - (t / 3) * 3;
  out[(size_t)s * 3 * LP1 + (size_t)r * LP1 + nnz[s]] = EOS_;
}

// ---------------- launcher ----------------
extern "C" void kernel_launch(void* const* d_in, const int* in_sizes, int n_in,
                              void* d_out, int out_size, void* d_ws, size_t ws_size,
                              hipStream_t stream) {
  (void)in_sizes; (void)n_in; (void)ws_size;
  const float* x = (const float*)d_in[0];
  int* out = (int*)d_out;

  // workspace carve-out (all 16B+ aligned)
  char* ws = (char*)d_ws;
  _Float16* afrag  = (_Float16*)ws;                                  //      8,192 B
  _Float16* blocks = (_Float16*)(ws + 8192);                         // 28,311,552 B
  int* zs     = (int*)(ws + 8192 + 28311552);                        // 56,623,104 B
  int* counts = (int*)(ws + 8192 + 28311552 + 56623104);             //     55,296 B
  int* offs   = counts + NCHUNKS;                                    //     55,296 B
  int* nnz    = offs + NCHUNKS;                                      //         32 B

  k_init_afrag<<<1, 256, 0, stream>>>(afrag);
  k_color<<<4608, 256, 0, stream>>>(x, blocks);            // 8*384*384 cells
  k_dct_wmma<<<6912, 256, 0, stream>>>(afrag, blocks, zs); // 24 planes * 288
  k_count<<<NCHUNKS, 256, 0, stream>>>(zs, counts);
  k_scan<<<NBATCH, 256, 0, stream>>>(counts, offs, nnz);
  int n4 = out_size / 4;
  k_fill<<<(n4 + 255) / 256, 256, 0, stream>>>(out, n4);
  k_scatter<<<NCHUNKS, 256, 0, stream>>>(zs, offs, out);
  k_eos<<<1, 32, 0, stream>>>(nnz, out);
}
